// GATLayer_15195594293512
// MI455X (gfx1250) — matrix operand order repaired and checked by reading, not compiled, not run on hardware
//
#include <hip/hip_runtime.h>
#include <math.h>

#define NNODES   50000
#define NEDGES   800000
#define INFEATS  128
#define DHEAD    16
#define NHEADS   8
#define NEG_SLOPE 0.2f

typedef __attribute__((ext_vector_type(16))) _Float16 v16h;
typedef __attribute__((ext_vector_type(8)))  float    v8f;

// float atomic-max via sign-split int/uint atomics (monotonic bit patterns)
__device__ __forceinline__ void atomicMaxF32(float* addr, float val) {
  if (val >= 0.0f) atomicMax((int*)addr, __float_as_int(val));
  else             atomicMin((unsigned int*)addr, __float_as_uint(val));
}

// ---- kernel 0: zero accumulator, init emax=-inf, denom=0 ----
__global__ void gat_init(float* __restrict__ out,
                         float* __restrict__ emax,
                         float* __restrict__ denom) {
  int i = blockIdx.x * blockDim.x + threadIdx.x;
  if (i < NNODES * INFEATS) out[i] = 0.0f;
  if (i < NNODES * NHEADS) {
    emax[i]  = -INFINITY;
    denom[i] = 0.0f;
  }
}

// ---- kernel 1: fh = feats @ W via WMMA; fused el/er head dot-products ----
// One block = one 16-row M tile; 8 waves = 8 heads (16-col N tiles).
__global__ void __launch_bounds__(256)
gat_proj(const float* __restrict__ feats,
         const float* __restrict__ W,
         const float* __restrict__ attn_l,
         const float* __restrict__ attn_r,
         float* __restrict__ fh,
         float* __restrict__ el,
         float* __restrict__ er) {
  const int lane = threadIdx.x & 31;
  const int head = threadIdx.x >> 5;           // 0..7
  const int m0   = blockIdx.x * 16;            // 3125 * 16 == 50000, no tail

  const int mrow = m0 + (lane & 15);           // A: row per lane
  const int khiA = (lane >> 4) << 3;           // A K offset: 0 or 8
  const int khiB = (lane >> 4) << 3;           // B K offset: 0 or 8
  const int ncol = head * DHEAD + (lane & 15); // global output column

  v8f c = {};
  #pragma unroll
  for (int kc = 0; kc < 4; ++kc) {             // K = 128 in 4 chunks of 32
    v16h a, b;
    const float* arow = feats + mrow * INFEATS + kc * 32 + khiA;
    #pragma unroll
    for (int i = 0; i < 16; ++i) {
      int k = (i & 7) + ((i >> 3) << 4);       // 0..7, 16..23 within chunk
      a[i] = (_Float16)arow[k];
    }
    const float* bcol = W + ncol;              // column of W, stride 128
    #pragma unroll
    for (int i = 0; i < 16; ++i) {
      int k = kc * 32 + khiB + (i & 7) + ((i >> 3) << 4);
      b[i] = (_Float16)bcol[k * (NHEADS * DHEAD)];
    }
    c = __builtin_amdgcn_wmma_f32_16x16x32_f16(false, a, false, b,
                                               (short)0, c, false, false);
  }

  const float al = attn_l[head * DHEAD + (lane & 15)];
  const float ar = attn_r[head * DHEAD + (lane & 15)];

  #pragma unroll
  for (int v = 0; v < 8; ++v) {
    const int row = m0 + v + ((lane >> 4) << 3);
    fh[row * INFEATS + ncol] = c[v];
    // el[row,head] = sum over the 16 columns of this head (lanes mod 16)
    float pl = c[v] * al;
    float pr = c[v] * ar;
    #pragma unroll
    for (int m = 1; m < 16; m <<= 1) {
      pl += __shfl_xor(pl, m, 32);
      pr += __shfl_xor(pr, m, 32);
    }
    if ((lane & 15) == 0) {
      el[row * NHEADS + head] = pl;
      er[row * NHEADS + head] = pr;
    }
  }
}

// ---- kernel 2: segment max of leaky-relu edge scores ----
__global__ void gat_edge_max(const float* __restrict__ el,
                             const float* __restrict__ er,
                             const int* __restrict__ src,
                             const int* __restrict__ dst,
                             float* __restrict__ emax) {
  int i = blockIdx.x * blockDim.x + threadIdx.x;
  if (i >= NEDGES * NHEADS) return;
  int e = i >> 3, h = i & 7;
  int s = src[e], d = dst[e];
  float x = el[s * NHEADS + h] + er[d * NHEADS + h];
  x = (x > 0.0f) ? x : NEG_SLOPE * x;
  atomicMaxF32(&emax[d * NHEADS + h], x);
}

// ---- kernel 3: softmax denominator ----
__global__ void gat_edge_den(const float* __restrict__ el,
                             const float* __restrict__ er,
                             const int* __restrict__ src,
                             const int* __restrict__ dst,
                             const float* __restrict__ emax,
                             float* __restrict__ denom) {
  int i = blockIdx.x * blockDim.x + threadIdx.x;
  if (i >= NEDGES * NHEADS) return;
  int e = i >> 3, h = i & 7;
  int s = src[e], d = dst[e];
  float x = el[s * NHEADS + h] + er[d * NHEADS + h];
  x = (x > 0.0f) ? x : NEG_SLOPE * x;
  atomicAdd(&denom[d * NHEADS + h], expf(x - emax[d * NHEADS + h]));
}

// ---- kernel 4: normalized attention coefficient per (edge, head) ----
// Computes exp once per (e,h) instead of 4x redundantly in the scatter.
__global__ void gat_edge_coef(const float* __restrict__ el,
                              const float* __restrict__ er,
                              const int* __restrict__ src,
                              const int* __restrict__ dst,
                              const float* __restrict__ emax,
                              const float* __restrict__ denom,
                              float* __restrict__ coef) {
  int i = blockIdx.x * blockDim.x + threadIdx.x;  // i == e*8 + h
  if (i >= NEDGES * NHEADS) return;
  int e = i >> 3, h = i & 7;
  int s = src[e], d = dst[e];
  float x = el[s * NHEADS + h] + er[d * NHEADS + h];
  x = (x > 0.0f) ? x : NEG_SLOPE * x;
  coef[i] = expf(x - emax[d * NHEADS + h]) / denom[d * NHEADS + h];
}

// ---- kernel 5: weighted scatter-sum; 4 features/thread, b128 gathers ----
__global__ void gat_scatter(const int* __restrict__ src,
                            const int* __restrict__ dst,
                            const float* __restrict__ coef,
                            const float* __restrict__ fh,
                            float* __restrict__ out) {
  int i = blockIdx.x * blockDim.x + threadIdx.x;  // NEDGES*32 = 25.6M
  if (i >= NEDGES * 32) return;
  int e = i >> 5;
  int q = (i & 31) << 2;          // feature base 0,4,...,124 (one head each)
  int h = q >> 4;
  int s = src[e], d = dst[e];
  float a = coef[e * NHEADS + h];
  const float4 f = *(const float4*)(fh + (size_t)s * INFEATS + q);
  float* o = out + (size_t)d * INFEATS + q;
  atomicAdd(o + 0, f.x * a);
  atomicAdd(o + 1, f.y * a);
  atomicAdd(o + 2, f.z * a);
  atomicAdd(o + 3, f.w * a);
}

// ---- kernel 6: residual + bias ----
__global__ void gat_final(const float* __restrict__ feats,
                          const float* __restrict__ bias,
                          float* __restrict__ out) {
  int i = blockIdx.x * blockDim.x + threadIdx.x;
  if (i >= NNODES * INFEATS) return;
  out[i] = out[i] + feats[i] + bias[i & 127];
}

extern "C" void kernel_launch(void* const* d_in, const int* in_sizes, int n_in,
                              void* d_out, int out_size, void* d_ws, size_t ws_size,
                              hipStream_t stream) {
  const float* feats  = (const float*)d_in[0];
  const float* W      = (const float*)d_in[1];
  const float* attn_l = (const float*)d_in[2];
  const float* attn_r = (const float*)d_in[3];
  const float* bias   = (const float*)d_in[4];
  const int*   src    = (const int*)d_in[5];
  const int*   dst    = (const int*)d_in[6];
  float* out = (float*)d_out;

  // workspace layout (floats): fh | el | er | emax | denom | coef  (~58 MB)
  float* fh    = (float*)d_ws;
  float* el    = fh    + (size_t)NNODES * INFEATS;
  float* er    = el    + (size_t)NNODES * NHEADS;
  float* emax  = er    + (size_t)NNODES * NHEADS;
  float* denom = emax  + (size_t)NNODES * NHEADS;
  float* coef  = denom + (size_t)NNODES * NHEADS;

  const int nf = NNODES * INFEATS;   // 6.4M
  const int eh = NEDGES * NHEADS;    // 6.4M
  const int ev = NEDGES * 32;        // 25.6M (4 features per thread)

  gat_init    <<<(nf + 255) / 256, 256, 0, stream>>>(out, emax, denom);
  gat_proj    <<<NNODES / 16,      256, 0, stream>>>(feats, W, attn_l, attn_r,
                                                     fh, el, er);
  gat_edge_max<<<(eh + 255) / 256, 256, 0, stream>>>(el, er, src, dst, emax);
  gat_edge_den<<<(eh + 255) / 256, 256, 0, stream>>>(el, er, src, dst, emax, denom);
  gat_edge_coef<<<(eh + 255) / 256, 256, 0, stream>>>(el, er, src, dst, emax,
                                                      denom, coef);
  gat_scatter <<<(ev + 255) / 256, 256, 0, stream>>>(src, dst, coef, fh, out);
  gat_final   <<<(nf + 255) / 256, 256, 0, stream>>>(feats, bias, out);
}
